// CausalSelfAttention_18923625906326
// MI455X (gfx1250) — matrix operand order repaired and checked by reading, not compiled, not run on hardware
//
#include <hip/hip_runtime.h>
#include <hip/hip_bf16.h>

// ---------------- types ----------------
typedef __attribute__((ext_vector_type(16))) _Float16 v16h;
typedef __attribute__((ext_vector_type(8)))  _Float16 v8h;
typedef __attribute__((ext_vector_type(8)))  float    v8f;
typedef __attribute__((ext_vector_type(4)))  int      v4i;

#define Bsz 2
#define Tsz 2048
#define Dsz 1024
#define Hn  16
#define HDsz 64

// ------------- CDNA5 async global->LDS (guarded; falls back to sync copy) ----
#if __has_builtin(__builtin_amdgcn_global_load_async_to_lds_b128) && \
    __has_builtin(__builtin_amdgcn_s_wait_asynccnt)
#define HAVE_ASYNC 1
#else
#define HAVE_ASYNC 0
#endif

#if HAVE_ASYNC
typedef __attribute__((address_space(1))) v4i* gptr_b128;  // global int4*
typedef __attribute__((address_space(3))) v4i* lptr_b128;  // LDS int4*
#endif

__device__ __forceinline__ void async_copy16(_Float16* l, const _Float16* g) {
#if HAVE_ASYNC
  __builtin_amdgcn_global_load_async_to_lds_b128((gptr_b128)(g), (lptr_b128)(l),
                                                 0, 0);
#else
  *(v8h*)l = *(const v8h*)g;
#endif
}

__device__ __forceinline__ void wait_async() {
#if HAVE_ASYNC
  __builtin_amdgcn_s_wait_asynccnt(0);
#endif
}

__device__ __forceinline__ v8f wmma_f16(v16h a, v16h b, v8f c) {
  return __builtin_amdgcn_wmma_f32_16x16x32_f16(false, a, false, b, (short)0, c,
                                                false, false);
}

// A-fragment (16x32 f16, row-major source, stride in halfs).
// lane<16: row=lane, k = {0..7, 16..23}; lane>=16: row=lane-16, k={8..15,24..31}
__device__ __forceinline__ v16h load_a_frag(const _Float16* base, int ld, int lane) {
  int row = lane & 15;
  int ko  = (lane < 16) ? 0 : 8;
  const _Float16* p = base + row * ld + ko;
  v16h f;
  *(v8h*)&f        = *(const v8h*)(p);
  *((v8h*)&f + 1)  = *(const v8h*)(p + 16);
  return f;
}

// B-fragment from a B^T tile (rows = N, cols = K contiguous, stride in halfs).
// lane<16: n=lane, k={0..7,16..23} ; lane>=16: n=lane-16, k={8..15,24..31}
__device__ __forceinline__ v16h load_bt_frag(const _Float16* base, int ld, int lane) {
  int col = lane & 15;
  int ko  = (lane < 16) ? 0 : 8;
  const _Float16* p = base + col * ld + ko;
  v16h f;
  *(v8h*)&f       = *(const v8h*)(p);
  *((v8h*)&f + 1) = *(const v8h*)(p + 16);
  return f;
}

// ---------------- fp32 -> f16 ----------------
__global__ void cvt_f32_f16(const float* __restrict__ in, _Float16* __restrict__ out, int n) {
  int i = blockIdx.x * blockDim.x + threadIdx.x;
  if (i < n) out[i] = (_Float16)in[i];
}

// ---------------- WMMA GEMM: C[M,N] = A[M,K] * B[K,N] ----------------
// block tile 128(M) x 128(N), 256 threads = 8 waves (4M x 2N), wave tile 32x64,
// K step 32: 8 wmma per wave per stage vs 6 fragment loads.
// mode 0: store fp32 into Cf.  mode 1: scatter f16 into Q/K/V [B,H,T,HD].
__global__ __launch_bounds__(256)
void gemm16(const _Float16* __restrict__ A, const _Float16* __restrict__ Bm,
            int M, int N, int K, float* Cf,
            _Float16* Qo, _Float16* Ko, _Float16* Vo, int mode) {
  __shared__ _Float16 As[128 * 40];    // [m][k] 32 K + 8 pad
  __shared__ _Float16 Bts[128 * 40];   // B^T tile: [n][k]
  const int tid  = threadIdx.x;
  const int lane = tid & 31;
  const int wave = tid >> 5;
  const int wm = wave & 3;            // 0..3 (M sub-tile of 32)
  const int wn = wave >> 2;           // 0..1 (N sub-tile of 64)
  const int mBase = blockIdx.y * 128;
  const int nBase = blockIdx.x * 128;

  v8f acc[2][4] = {};

  for (int k0 = 0; k0 < K; k0 += 32) {
    __syncthreads();
    {
      // A tile: 128 rows x 32 halfs; 32B per thread via 2x async b128
      int arow = tid >> 1, ac = (tid & 1) * 16;
      const _Float16* ga = A + (size_t)(mBase + arow) * K + k0 + ac;
      async_copy16(&As[arow * 40 + ac], ga);
      async_copy16(&As[arow * 40 + ac + 8], ga + 8);
      // B tile (32 x 128) -> transposed into LDS
#pragma unroll
      for (int c = tid; c < 512; c += 256) {
        int kb = c >> 4, nc = (c & 15) * 8;
        v8h bv = *(const v8h*)(Bm + (size_t)(k0 + kb) * N + nBase + nc);
#pragma unroll
        for (int j = 0; j < 8; ++j) Bts[(nc + j) * 40 + kb] = bv[j];
      }
    }
    wait_async();
    __syncthreads();

    v16h a0 = load_a_frag(As + (wm * 32 + 0)  * 40, 40, lane);
    v16h a1 = load_a_frag(As + (wm * 32 + 16) * 40, 40, lane);
    v16h b[4];
#pragma unroll
    for (int ni = 0; ni < 4; ++ni)
      b[ni] = load_bt_frag(Bts + (wn * 64 + ni * 16) * 40, 40, lane);
#pragma unroll
    for (int ni = 0; ni < 4; ++ni) {
      acc[0][ni] = wmma_f16(a0, b[ni], acc[0][ni]);
      acc[1][ni] = wmma_f16(a1, b[ni], acc[1][ni]);
    }
  }

#pragma unroll
  for (int mi = 0; mi < 2; ++mi)
#pragma unroll
    for (int ni = 0; ni < 4; ++ni)
#pragma unroll
      for (int e = 0; e < 8; ++e) {
        int row = mBase + wm * 32 + mi * 16 + ((lane < 16) ? e : e + 8);
        int col = nBase + wn * 64 + ni * 16 + (lane & 15);
        float v = acc[mi][ni][e];
        if (mode == 0) {
          Cf[(size_t)row * N + col] = v;
        } else {
          int which = col >> 10;          // 0=q 1=k 2=v   (D=1024)
          int rem   = col & 1023;
          int hh = rem >> 6, dd = rem & 63;
          int bb = row >> 11, tt = row & 2047;   // T=2048
          _Float16* dst = (which == 0) ? Qo : ((which == 1) ? Ko : Vo);
          dst[(((size_t)bb * Hn + hh) * Tsz + tt) * HDsz + dd] = (_Float16)v;
        }
      }
}

// ---------------- RoPE on Q,K in [B,H,T,HD] ----------------
__global__ void rope_qk(_Float16* __restrict__ Q, _Float16* __restrict__ K) {
  int i = blockIdx.x * blockDim.x + threadIdx.x;   // B*H*T*32 threads
  int d2 = i & 31;
  int t  = (i >> 5) & 2047;
  int bh = i >> 16;
  size_t base = ((size_t)bh * Tsz + t) * HDsz;
  // inv_freq = 10000^(-d2/32);  ln(10000)/32 = 0.28782313662
  float inv = __expf(-(float)d2 * 0.28782313662425572f);
  float f = (float)t * inv;
  float cs = __cosf(f), sn = __sinf(f);
  float a0 = (float)Q[base + d2], a1 = (float)Q[base + d2 + 32];
  Q[base + d2]      = (_Float16)(a0 * cs - a1 * sn);
  Q[base + d2 + 32] = (_Float16)(a1 * cs + a0 * sn);
  float b0 = (float)K[base + d2], b1 = (float)K[base + d2 + 32];
  K[base + d2]      = (_Float16)(b0 * cs - b1 * sn);
  K[base + d2 + 32] = (_Float16)(b1 * cs + b0 * sn);
}

// ---------------- Flash attention ----------------
// grid: (T/64, B*H); block: 128 (4 waves). Each wave owns 16 query rows.
__global__ __launch_bounds__(128)
void flash_attn(const _Float16* __restrict__ Qb, const _Float16* __restrict__ Kb,
                const _Float16* __restrict__ Vb, _Float16* __restrict__ Ah) {
  __shared__ _Float16 Qs[64 * 72];
  __shared__ _Float16 Ks[64 * 72];    // K tile row-major == B^T layout for Q*K^T
  __shared__ _Float16 VTs[64 * 72];   // V^T tile: [hd][key]
  __shared__ _Float16 Ps[4 * 16 * 72];

  const int tid = threadIdx.x, lane = tid & 31, wave = tid >> 5;
  const int bh = blockIdx.y;
  const int bb = bh >> 4, hh = bh & 15;
  const int q0 = blockIdx.x * 64;
  const _Float16* Qp = Qb + (size_t)bh * Tsz * HDsz;
  const _Float16* Kp = Kb + (size_t)bh * Tsz * HDsz;
  const _Float16* Vp = Vb + (size_t)bh * Tsz * HDsz;

  // load Q tile (64 x 64) via async DMA to LDS
  for (int c = tid; c < 512; c += 128) {
    int row = c >> 3, ch = (c & 7) * 8;
    async_copy16(&Qs[row * 72 + ch], &Qp[(size_t)(q0 + row) * HDsz + ch]);
  }

  v8f o[4] = {};
  float mrow[8], lrow[8];
#pragma unroll
  for (int e = 0; e < 8; ++e) { mrow[e] = -1e30f; lrow[e] = 0.f; }

  const int rbase = q0 + wave * 16;
  const int ktmax = (q0 + 63) >> 6;

  for (int kt = 0; kt <= ktmax; ++kt) {
    __syncthreads();
    for (int c = tid; c < 512; c += 128) {            // K tile: async DMA
      int row = c >> 3, ch = (c & 7) * 8;
      async_copy16(&Ks[row * 72 + ch], &Kp[(size_t)(kt * 64 + row) * HDsz + ch]);
    }
    for (int c = tid; c < 512; c += 128) {            // V tile, transposed
      int row = c >> 3, ch = (c & 7) * 8;
      v8h tv = *(const v8h*)&Vp[(size_t)(kt * 64 + row) * HDsz + ch];
#pragma unroll
      for (int j = 0; j < 8; ++j) VTs[(ch + j) * 72 + row] = tv[j];
    }
    wait_async();
    __syncthreads();

    // S = Q * K^T  (16 x 64 per wave)
    v8f s[4] = {};
#pragma unroll
    for (int kk = 0; kk < 2; ++kk) {
      v16h aq = load_a_frag(Qs + (wave * 16) * 72 + kk * 32, 72, lane);
#pragma unroll
      for (int n = 0; n < 4; ++n) {
        v16h bk = load_bt_frag(Ks + (n * 16) * 72 + kk * 32, 72, lane);
        s[n] = wmma_f16(aq, bk, s[n]);
      }
    }

    // mask + online softmax (rows live in vector elems; 16-lane halves share a row)
    float mt[8];
#pragma unroll
    for (int e = 0; e < 8; ++e) {
      int r = rbase + ((lane < 16) ? e : e + 8);
      float mx = -1e30f;
#pragma unroll
      for (int n = 0; n < 4; ++n) {
        int kg = kt * 64 + n * 16 + (lane & 15);
        float val = s[n][e] * 0.125f;                 // HD^-0.5
        if (kg > r) val = -1e30f;
        s[n][e] = val;
        mx = fmaxf(mx, val);
      }
      mx = fmaxf(mx, __shfl_xor(mx, 1, 32));
      mx = fmaxf(mx, __shfl_xor(mx, 2, 32));
      mx = fmaxf(mx, __shfl_xor(mx, 4, 32));
      mx = fmaxf(mx, __shfl_xor(mx, 8, 32));
      mt[e] = mx;
    }
#pragma unroll
    for (int e = 0; e < 8; ++e) {
      float mnew  = fmaxf(mrow[e], mt[e]);
      float alpha = __expf(mrow[e] - mnew);
      mrow[e] = mnew;
      float ls = 0.f;
#pragma unroll
      for (int n = 0; n < 4; ++n) {
        float p = __expf(s[n][e] - mnew);
        s[n][e] = p;
        ls += p;
      }
      ls += __shfl_xor(ls, 1, 32);
      ls += __shfl_xor(ls, 2, 32);
      ls += __shfl_xor(ls, 4, 32);
      ls += __shfl_xor(ls, 8, 32);
      lrow[e] = lrow[e] * alpha + ls;
#pragma unroll
      for (int n = 0; n < 4; ++n) o[n][e] *= alpha;
    }

    // stage P (D-layout -> LDS row-major -> A-layout), wave-private region
    _Float16* Pw = Ps + wave * 16 * 72;
#pragma unroll
    for (int n = 0; n < 4; ++n)
#pragma unroll
      for (int e = 0; e < 8; ++e) {
        int r = (lane < 16) ? e : e + 8;
        Pw[r * 72 + n * 16 + (lane & 15)] = (_Float16)s[n][e];
      }

    // O += P * V
#pragma unroll
    for (int kk = 0; kk < 2; ++kk) {
      v16h ap = load_a_frag(Pw + kk * 32, 72, lane);
#pragma unroll
      for (int n = 0; n < 4; ++n) {
        v16h bv = load_bt_frag(VTs + (n * 16) * 72 + kk * 32, 72, lane);
        o[n] = wmma_f16(ap, bv, o[n]);
      }
    }
  }

  // normalize + store to attn buffer [B*T, D] f16 (A of the out-proj GEMM)
#pragma unroll
  for (int n = 0; n < 4; ++n)
#pragma unroll
    for (int e = 0; e < 8; ++e) {
      int r = (lane < 16) ? e : e + 8;
      int t = q0 + wave * 16 + r;
      float val = o[n][e] / lrow[e];
      Ah[((size_t)bb * Tsz + t) * Dsz + hh * HDsz + n * 16 + (lane & 15)] =
          (_Float16)val;
    }
}

// ---------------- launch ----------------
extern "C" void kernel_launch(void* const* d_in, const int* in_sizes, int n_in,
                              void* d_out, int out_size, void* d_ws, size_t ws_size,
                              hipStream_t stream) {
  const float* x    = (const float*)d_in[0];   // [B,T,D]
  const float* wqkv = (const float*)d_in[1];   // [D,3D]
  const float* wout = (const float*)d_in[2];   // [D,D]
  float* out = (float*)d_out;                  // [B,T,D] fp32

  const int nX = Bsz * Tsz * Dsz;              // 4,194,304
  const int nWqkv = Dsz * 3 * Dsz;             // 3,145,728
  const int nWout = Dsz * Dsz;                 // 1,048,576
  const int nHead = Bsz * Hn * Tsz * HDsz;     // 4,194,304

  _Float16* Xh    = (_Float16*)d_ws;
  _Float16* Wqkvh = Xh + nX;
  _Float16* Wouth = Wqkvh + nWqkv;
  _Float16* Qb    = Wouth + nWout;
  _Float16* Kb    = Qb + nHead;
  _Float16* Vb    = Kb + nHead;
  _Float16* Ahb   = Vb + nHead;                // attn output f16 [B*T, D]

  cvt_f32_f16<<<(nX + 255) / 256, 256, 0, stream>>>(x, Xh, nX);
  cvt_f32_f16<<<(nWqkv + 255) / 256, 256, 0, stream>>>(wqkv, Wqkvh, nWqkv);
  cvt_f32_f16<<<(nWout + 255) / 256, 256, 0, stream>>>(wout, Wouth, nWout);

  // QKV projection: [4096,1024] x [1024,3072] -> scatter Q/K/V
  gemm16<<<dim3(3 * Dsz / 128, Bsz * Tsz / 128), 256, 0, stream>>>(
      Xh, Wqkvh, Bsz * Tsz, 3 * Dsz, Dsz, nullptr, Qb, Kb, Vb, 1);

  rope_qk<<<(Bsz * Hn * Tsz * 32) / 256, 256, 0, stream>>>(Qb, Kb);

  flash_attn<<<dim3(Tsz / 64, Bsz * Hn), 128, 0, stream>>>(Qb, Kb, Vb, Ahb);

  // out projection: [4096,1024] x [1024,1024] -> fp32 d_out
  gemm16<<<dim3(Dsz / 128, Bsz * Tsz / 128), 256, 0, stream>>>(
      Ahb, Wouth, Bsz * Tsz, Dsz, Dsz, out, nullptr, nullptr, nullptr, 0);
}
